// CalibrationAttention_6640019439786
// MI455X (gfx1250) — matrix operand order repaired and checked by reading, not compiled, not run on hardware
//
#include <hip/hip_runtime.h>
#include <hip/hip_bf16.h>
#include <cstdint>
#include <cstddef>

// ---- problem dims ----
#define NB    16
#define NTOK  577
#define CDIM  768
#define NH    12
#define DH    64
#define HIDD  384
#define C3    (3*CDIM)
#define MTOT  (NB*NTOK)        // 9232
#define QTILES ((NTOK+15)/16)  // 37
#define MBLK  64
#define MGRID ((MTOT + MBLK - 1) / MBLK)  // 145

typedef __attribute__((ext_vector_type(16))) __bf16 v16bf;
typedef __attribute__((ext_vector_type(8)))  float  v8f;
typedef __attribute__((ext_vector_type(4)))  int    v4i;

union FragBF { uint4 u[2]; v16bf v; unsigned short us[16]; };

// ---- CDNA5 async global->LDS path (ASYNCcnt), guarded for toolchain ----
#if defined(__has_builtin)
#if __has_builtin(__builtin_amdgcn_global_load_async_to_lds_b128)
#define HAVE_ASYNC_LDS 1
#endif
#endif

typedef __attribute__((address_space(1))) v4i* as1_v4i_p;
typedef __attribute__((address_space(3))) v4i* as3_v4i_p;

static __device__ __forceinline__ void async_g2l_b128(const void* g, void* l) {
#ifdef HAVE_ASYNC_LDS
  __builtin_amdgcn_global_load_async_to_lds_b128((as1_v4i_p)g, (as3_v4i_p)l, 0, 0);
#else
  *(uint4*)l = *(const uint4*)g;
#endif
}
static __device__ __forceinline__ void async_wait0() {
#ifdef HAVE_ASYNC_LDS
#if __has_builtin(__builtin_amdgcn_s_wait_asynccnt)
  __builtin_amdgcn_s_wait_asynccnt(0);
#else
  asm volatile("s_wait_asynccnt 0x0" ::: "memory");
#endif
#endif
}

static __device__ __forceinline__ __bf16 f2bf(float f) {
  unsigned u = __builtin_bit_cast(unsigned, f);
  unsigned r = u + 0x7FFFu + ((u >> 16) & 1u);   // round-to-nearest-even
  unsigned short s = (unsigned short)(r >> 16);
  return __builtin_bit_cast(__bf16, s);
}
static __device__ __forceinline__ __bf16 us2bf(unsigned short s) {
  return __builtin_bit_cast(__bf16, s);
}
static __device__ __forceinline__ v8f wmma_bf16(v16bf a, v16bf b, v8f c) {
  return __builtin_amdgcn_wmma_f32_16x16x32_bf16(false, a, false, b, (short)0, c, false, false);
}
static __device__ __forceinline__ float rmax16(float v) {
  v = fmaxf(v, __shfl_xor(v, 1, 32));
  v = fmaxf(v, __shfl_xor(v, 2, 32));
  v = fmaxf(v, __shfl_xor(v, 4, 32));
  v = fmaxf(v, __shfl_xor(v, 8, 32));
  return v;
}
static __device__ __forceinline__ float rsum16(float v) {
  v += __shfl_xor(v, 1, 32);
  v += __shfl_xor(v, 2, 32);
  v += __shfl_xor(v, 4, 32);
  v += __shfl_xor(v, 8, 32);
  return v;
}

// ---------------- fp32 -> bf16 conversion ----------------
__global__ void cvt_bf16_kernel(const float* __restrict__ in,
                                __bf16* __restrict__ out, size_t n) {
  size_t i = (size_t)blockIdx.x * blockDim.x + threadIdx.x;
  size_t stride = (size_t)gridDim.x * blockDim.x;
  for (; i < n; i += stride) out[i] = f2bf(in[i]);
}

// ---------------- temperature MLP (tiny) ----------------
__global__ __launch_bounds__(256)
void temp_kernel(const float* __restrict__ x,
                 const float* __restrict__ w1, const float* __restrict__ b1,
                 const float* __restrict__ w2, const float* __restrict__ b2,
                 float* __restrict__ temp) {
  __shared__ float cls_s[CDIM];
  __shared__ float red[256];
  int b = blockIdx.x, tid = threadIdx.x;
  const float* cls = x + (size_t)b * NTOK * CDIM;  // token 0
  for (int i = tid; i < CDIM; i += 256) cls_s[i] = cls[i];
  __syncthreads();
  float part = 0.f;
  for (int j = tid; j < HIDD; j += 256) {
    float acc = b1[j];
    for (int k = 0; k < CDIM; ++k) acc += cls_s[k] * w1[k * HIDD + j];
    part += fmaxf(acc, 0.f) * w2[j];
  }
  red[tid] = part;
  __syncthreads();
  for (int s = 128; s > 0; s >>= 1) {
    if (tid < s) red[tid] += red[tid + s];
    __syncthreads();
  }
  if (tid == 0) {
    float t = red[0] + b2[0];
    float sig = 1.f / (1.f + __expf(-t));
    temp[b] = 0.5f + 2.5f * sig;
  }
}

// ------- bf16 WMMA GEMM: block tile 64x128, 8 waves, 4 WMMA/wave/K-step -------
// MODE 0: epilogue scatters bf16 into q/k/v [B][H][N][D]
// MODE 1: epilogue adds bias, writes fp32 row-major [M][NCOL]
template <int MODE>
__global__ __launch_bounds__(256)
void gemm_bf16_kernel(const __bf16* __restrict__ A, const __bf16* __restrict__ Bm,
                      int K, int NCOL,
                      float* __restrict__ outf, const float* __restrict__ bias,
                      __bf16* __restrict__ qb, __bf16* __restrict__ kb,
                      __bf16* __restrict__ vb) {
  __shared__ __bf16 As[64][40];     // row-major A tile, 80B rows (16B aligned)
  __shared__ __bf16 Bt[128][40];    // B tile stored transposed [n][k]
  int tid = threadIdx.x;
  int wave = tid >> 5, lane = tid & 31, half = lane >> 4, r16 = lane & 15;
  int wm = wave >> 1, wn = wave & 1;   // wave tile: rows wm*16, cols wn*64
  int m0 = blockIdx.y * MBLK;
  int n0 = blockIdx.x * 128;

  v8f c[4];
  #pragma unroll
  for (int j = 0; j < 4; ++j) c[j] = (v8f){0.f,0.f,0.f,0.f,0.f,0.f,0.f,0.f};

  for (int k0 = 0; k0 < K; k0 += 32) {
    // stage A 64x32 via async global->LDS: 256 x b128, one chunk per thread
    {
      int row = tid >> 2, cc = tid & 3;
      int rg = m0 + row; rg = (rg < MTOT) ? rg : (MTOT - 1);  // clamp, masked later
      async_g2l_b128(A + (size_t)rg * K + k0 + cc * 8, &As[row][cc * 8]);
    }
    // stage B 32x128 transposed (needs transpose -> VGPR scatter path)
    #pragma unroll
    for (int i = 0; i < 2; ++i) {
      int idx = tid + i * 256;           // 512 uint4 total
      int krow = idx >> 4, c4 = idx & 15;
      uint4 g = *(const uint4*)(Bm + (size_t)(k0 + krow) * NCOL + n0 + c4 * 8);
      const unsigned short* gs = (const unsigned short*)&g;
      #pragma unroll
      for (int j = 0; j < 8; ++j) Bt[c4 * 8 + j][krow] = us2bf(gs[j]);
    }
    async_wait0();
    __syncthreads();

    // one A fragment reused across 4 B tiles
    FragBF a;
    a.u[0] = *(const uint4*)&As[wm * 16 + r16][half * 16];
    a.u[1] = *(const uint4*)&As[wm * 16 + r16][half * 16 + 8];
    #pragma unroll
    for (int j = 0; j < 4; ++j) {
      FragBF bf;
      bf.u[0] = *(const uint4*)&Bt[wn * 64 + j * 16 + r16][half * 16];
      bf.u[1] = *(const uint4*)&Bt[wn * 64 + j * 16 + r16][half * 16 + 8];
      c[j] = wmma_bf16(a.v, bf.v, c[j]);
    }
    __syncthreads();
  }

  #pragma unroll
  for (int j = 0; j < 4; ++j) {
    int col = n0 + wn * 64 + j * 16 + r16;
    #pragma unroll
    for (int r = 0; r < 8; ++r) {
      int row = m0 + wm * 16 + half * 8 + r;   // C/D layout: row = half*8 + vgpr
      if (row >= MTOT) continue;
      float val = c[j][r];
      if (MODE == 0) {
        int s = col / CDIM;
        int rem = col - s * CDIM;
        int h = rem >> 6, d = rem & 63;
        int bb = row / NTOK, n = row - bb * NTOK;
        size_t off = (((size_t)bb * NH + h) * NTOK + n) * DH + d;
        __bf16* dst = (s == 0) ? qb : (s == 1) ? kb : vb;
        dst[off] = f2bf(val);
      } else {
        outf[(size_t)row * NCOL + col] = val + bias[col];
      }
    }
  }
}

// ---------------- flash attention: 1 wave per (b,h,qtile) ----------------
__global__ __launch_bounds__(32)
void attn_kernel(const __bf16* __restrict__ q, const __bf16* __restrict__ k,
                 const __bf16* __restrict__ v, const float* __restrict__ temp,
                 __bf16* __restrict__ ao) {
  __shared__ __bf16 Qs[16][72];   // [qrow][d], 144B rows
  __shared__ __bf16 Ks[16][72];   // [key][d]
  __shared__ __bf16 Vst[64][24];  // transposed: [d][key], 48B rows
  __shared__ __bf16 Ps[16][24];   // [qrow][key]

  int lane = threadIdx.x, half = lane >> 4, r16 = lane & 15;
  int qt = blockIdx.x % QTILES;
  int bh = blockIdx.x / QTILES;
  int h = bh % NH, b = bh / NH;
  const __bf16* qp = q + (size_t)bh * NTOK * DH;
  const __bf16* kp = k + (size_t)bh * NTOK * DH;
  const __bf16* vp = v + (size_t)bh * NTOK * DH;
  float inv = 0.125f / temp[b];   // D^-0.5 / temperature

  // stage Q tile via async (rows clamped; tail rows discarded at store)
  #pragma unroll
  for (int i = 0; i < 2; ++i) {
    int idx = lane + i * 32;           // 64 b128 chunks
    int row = idx >> 2, cc = idx & 3;
    int qrow = qt * 16 + row; qrow = (qrow < NTOK) ? qrow : (NTOK - 1);
    async_g2l_b128(qp + (size_t)qrow * DH + cc * 8, &Qs[row][cc * 8]);
  }
  async_wait0();
  __syncthreads();

  FragBF aq[2];
  #pragma unroll
  for (int cc = 0; cc < 2; ++cc) {
    aq[cc].u[0] = *(const uint4*)&Qs[r16][cc * 32 + half * 16];
    aq[cc].u[1] = *(const uint4*)&Qs[r16][cc * 32 + half * 16 + 8];
  }

  float mrow[8], lrow[8];
  v8f acc[4];
  #pragma unroll
  for (int r = 0; r < 8; ++r) { mrow[r] = -1e30f; lrow[r] = 0.f; }
  #pragma unroll
  for (int nt = 0; nt < 4; ++nt) acc[nt] = (v8f){0.f,0.f,0.f,0.f,0.f,0.f,0.f,0.f};

  for (int kt = 0; kt < QTILES; ++kt) {
    // K tile: async row-major; V tile: transposed scatter (clamped rows; the
    // softmax mask zeroes P for key >= NTOK so clamped K/V rows contribute 0)
    #pragma unroll
    for (int i = 0; i < 2; ++i) {
      int idx = lane + i * 32;
      int row = idx >> 2, cc = idx & 3;
      int krow = kt * 16 + row; krow = (krow < NTOK) ? krow : (NTOK - 1);
      async_g2l_b128(kp + (size_t)krow * DH + cc * 8, &Ks[row][cc * 8]);
    }
    #pragma unroll
    for (int i = 0; i < 4; ++i) {
      int idx = lane + i * 32;
      int row = idx >> 3, c8 = idx & 7;
      int krow = kt * 16 + row; krow = (krow < NTOK) ? krow : (NTOK - 1);
      uint4 vv = *(const uint4*)(vp + (size_t)krow * DH + c8 * 8);
      const unsigned short* vs = (const unsigned short*)&vv;
      #pragma unroll
      for (int j = 0; j < 8; ++j) Vst[c8 * 8 + j][row] = us2bf(vs[j]);
    }
    async_wait0();
    __syncthreads();

    // S = Q * K^T (K-dim = DH = 64 -> two WMMAs)
    v8f s = {0.f,0.f,0.f,0.f,0.f,0.f,0.f,0.f};
    #pragma unroll
    for (int cc = 0; cc < 2; ++cc) {
      FragBF bk;   // B(d, key): elem i = Ks[key=col][d = cc*32 + half*16 + i]
      bk.u[0] = *(const uint4*)&Ks[r16][cc * 32 + half * 16];
      bk.u[1] = *(const uint4*)&Ks[r16][cc * 32 + half * 16 + 8];
      s = wmma_bf16(aq[cc].v, bk.v, s);
    }

    // scale + tail mask
    int kidx = kt * 16 + r16;
    bool kvalid = (kidx < NTOK);
    #pragma unroll
    for (int r = 0; r < 8; ++r) s[r] = kvalid ? s[r] * inv : -1e30f;

    // online softmax (each row lives in 16 lanes of one wave half)
    #pragma unroll
    for (int r = 0; r < 8; ++r) {
      float mx = rmax16(s[r]);
      float mn = fmaxf(mrow[r], mx);
      float p  = __expf(s[r] - mn);
      float ps = rsum16(p);
      float alpha = __expf(mrow[r] - mn);
      lrow[r] = lrow[r] * alpha + ps;
      mrow[r] = mn;
      #pragma unroll
      for (int nt = 0; nt < 4; ++nt) acc[nt][r] *= alpha;
      Ps[half * 8 + r][r16] = f2bf(p);
    }
    __syncthreads();

    // O += P * V  (K padded 16 -> 32 with zeros; A half-1 lanes = 0)
    FragBF ap;
    ap.u[0] = (half == 0) ? *(const uint4*)&Ps[r16][0] : (uint4){0u,0u,0u,0u};
    ap.u[1] = (half == 0) ? *(const uint4*)&Ps[r16][8] : (uint4){0u,0u,0u,0u};
    #pragma unroll
    for (int nt = 0; nt < 4; ++nt) {
      FragBF bv;
      if (half == 0) {  // keys 0..15 real; keys 16..31 are the zero pad
        bv.u[0] = *(const uint4*)&Vst[nt * 16 + r16][0];
        bv.u[1] = *(const uint4*)&Vst[nt * 16 + r16][8];
      } else {
        bv.u[0] = (uint4){0u,0u,0u,0u};
        bv.u[1] = (uint4){0u,0u,0u,0u};
      }
      acc[nt] = wmma_bf16(ap.v, bv.v, acc[nt]);
    }
    __syncthreads();
  }

  // write O/l as bf16 into [B][N][C] for the proj GEMM
  #pragma unroll
  for (int r = 0; r < 8; ++r) {
    int nrow = qt * 16 + half * 8 + r;
    if (nrow < NTOK) {
      float linv = 1.f / lrow[r];
      #pragma unroll
      for (int nt = 0; nt < 4; ++nt) {
        int d = nt * 16 + r16;
        ao[((size_t)b * NTOK + nrow) * CDIM + h * DH + d] = f2bf(acc[nt][r] * linv);
      }
    }
  }
}

// ---------------- host-side launcher ----------------
extern "C" void kernel_launch(void* const* d_in, const int* in_sizes, int n_in,
                              void* d_out, int out_size, void* d_ws, size_t ws_size,
                              hipStream_t stream) {
  (void)in_sizes; (void)n_in; (void)out_size; (void)ws_size;
  const float* x      = (const float*)d_in[0];
  const float* qkv_w  = (const float*)d_in[1];
  const float* proj_w = (const float*)d_in[2];
  const float* proj_b = (const float*)d_in[3];
  const float* t_w1   = (const float*)d_in[4];
  const float* t_b1   = (const float*)d_in[5];
  const float* t_w2   = (const float*)d_in[6];
  const float* t_b2   = (const float*)d_in[7];
  float* out = (float*)d_out;

  // workspace carve-up (256B aligned slabs), total ~75.6 MB
  char* ws = (char*)d_ws;
  size_t off = 0;
  float*  temp   = (float*)(ws + off);  off += 256;
  __bf16* xb     = (__bf16*)(ws + off); off += (size_t)MTOT * CDIM * 2;
  __bf16* wqkvb  = (__bf16*)(ws + off); off += (size_t)CDIM * C3  * 2;
  __bf16* wprojb = (__bf16*)(ws + off); off += (size_t)CDIM * CDIM * 2;
  __bf16* qb     = (__bf16*)(ws + off); off += (size_t)MTOT * CDIM * 2;
  __bf16* kb     = (__bf16*)(ws + off); off += (size_t)MTOT * CDIM * 2;
  __bf16* vb     = (__bf16*)(ws + off); off += (size_t)MTOT * CDIM * 2;
  __bf16* aob    = (__bf16*)(ws + off); off += (size_t)MTOT * CDIM * 2;

  cvt_bf16_kernel<<<2048, 256, 0, stream>>>(x, xb, (size_t)MTOT * CDIM);
  cvt_bf16_kernel<<<2048, 256, 0, stream>>>(qkv_w, wqkvb, (size_t)CDIM * C3);
  cvt_bf16_kernel<<<512, 256, 0, stream>>>(proj_w, wprojb, (size_t)CDIM * CDIM);

  temp_kernel<<<NB, 256, 0, stream>>>(x, t_w1, t_b1, t_w2, t_b2, temp);

  // QKV: M=9232, N=2304, K=768
  gemm_bf16_kernel<0><<<dim3(C3 / 128, MGRID), 256, 0, stream>>>(
      xb, wqkvb, CDIM, C3, nullptr, nullptr, qb, kb, vb);

  // attention: one wave per (b, h, 16-query tile)
  attn_kernel<<<NB * NH * QTILES, 32, 0, stream>>>(qb, kb, vb, temp, aob);

  // proj: M=9232, N=768, K=768, + bias, fp32 out
  gemm_bf16_kernel<1><<<dim3(CDIM / 128, MGRID), 256, 0, stream>>>(
      aob, wprojb, CDIM, CDIM, out, proj_b, nullptr, nullptr, nullptr);
}
